// ModulatedConv2d_25013889532227
// MI455X (gfx1250) — compile-verified, matrix operand order checked by reading, not run on hardware
//
#include <hip/hip_runtime.h>

// ---------------- problem constants ----------------
#define B_     16
#define CI_    512
#define CO_    512
#define HW_    64
#define P_     4096          // 64*64 spatial
#define EPS_   1e-8f

typedef __bf16 v16bf  __attribute__((ext_vector_type(16)));
typedef __bf16 bf16x8 __attribute__((ext_vector_type(8)));
typedef float  v8f    __attribute__((ext_vector_type(8)));
typedef unsigned int u32x4 __attribute__((ext_vector_type(4)));
typedef int    i32x8  __attribute__((ext_vector_type(8)));
typedef int    i32x4  __attribute__((ext_vector_type(4)));

// -------- kernel 1: s = style @ (affine_w/sqrt(512)).T + affine_b --------
__global__ __launch_bounds__(256)
void style_affine_k(const float* __restrict__ style, const float* __restrict__ aw,
                    const float* __restrict__ ab, float* __restrict__ s)
{
    int idx = blockIdx.x * blockDim.x + threadIdx.x;      // b*512 + i
    int b = idx >> 9, i = idx & 511;
    const float scale = 0.044194173824159216f;            // 1/sqrt(512)
    const float* st = style + b * 512;
    const float* w  = aw    + i * 512;
    float acc = 0.f;
    #pragma unroll 8
    for (int j = 0; j < 512; ++j) acc += st[j] * w[j];
    s[idx] = acc * scale + ab[i];
}

// -------- kernel 2: bf16 weight repack [tap][Co][Ci] + wsq = sum_kk w^2 --------
__global__ __launch_bounds__(256)
void weight_prep_k(const float* __restrict__ weight, __bf16* __restrict__ wb,
                   float* __restrict__ wsq)
{
    int idx = blockIdx.x * blockDim.x + threadIdx.x;      // o*512 + i
    int o = idx >> 9, i = idx & 511;
    float sum = 0.f;
    #pragma unroll
    for (int t = 0; t < 9; ++t) {
        float v = weight[(size_t)(o * 512 + i) * 9 + t];
        sum += v * v;
        wb[(size_t)t * (CO_ * CI_) + o * 512 + i] = (__bf16)v;
    }
    wsq[idx] = sum;
}

// -------- kernel 3: sigma_inv[b,o] = rsqrt(sum_i wsq[o,i]*s[b,i]^2 + eps) --------
__global__ __launch_bounds__(256)
void sigma_k(const float* __restrict__ wsq, const float* __restrict__ s,
             float* __restrict__ sig)
{
    int idx = blockIdx.x * blockDim.x + threadIdx.x;      // b*512 + o
    int b = idx >> 9, o = idx & 511;
    const float* wr = wsq + o * 512;
    const float* sr = s   + b * 512;
    float acc = 0.f;
    #pragma unroll 8
    for (int i = 0; i < 512; ++i) { float sv = sr[i]; acc += wr[i] * sv * sv; }
    sig[idx] = rsqrtf(acc + EPS_);
}

// -------- LDS helpers --------
typedef __attribute__((address_space(3))) const void* lds_cptr_t;

__device__ __forceinline__ unsigned int lds_addr_of(const void* p)
{
    // generic -> LDS addrspace cast recovers the 32-bit LDS byte offset
    return (unsigned int)(unsigned long long)(lds_cptr_t)p;
}

__device__ __forceinline__ v16bf load_frag16(const __bf16* base)
{
    bf16x8 lo = *(const bf16x8*)(base);
    bf16x8 hi = *(const bf16x8*)(base + 16);
    v16bf r;
    #pragma unroll
    for (int i = 0; i < 8; ++i) { r[i] = lo[i]; r[i + 8] = hi[i]; }
    return r;
}

// -------- TDM: A tile (128 x 32 bf16, row pitch 40 elems via LDS padding) --------
// D# per CDNA5 ISA: 2D tile, data_size=2B, tile_dim0=32, tile_dim1=128,
// tensor_dim0=tensor_dim1=512, dim0_stride=512 elems.
// pad_enable, pad_interval=16 DWORDs (enc 3), pad_amount=4 DWORDs (enc 3)
// -> 64B data + 16B pad per row == 80B pitch (APITCH=40 bf16).
__device__ __forceinline__ void tdm_load_A(const __bf16* gsrc, unsigned int lds_byte_off)
{
    unsigned long long ga = (unsigned long long)gsrc;
    u32x4 g0;
    g0[0] = 1u;                                            // count=1 (valid), no gather
    g0[1] = lds_byte_off;                                  // lds_addr
    g0[2] = (unsigned int)ga;                              // global_addr[31:0]
    g0[3] = (unsigned int)((ga >> 32) & 0x1FFFFFFull)      // global_addr[56:32]
            | (2u << 30);                                  // type=2 (image)
    i32x8 g1;
    g1[0] = (int)((1u << 16)                               // data_size = 2B
                | (1u << 20)                               // pad_enable
                | (3u << 22)                               // pad_interval: 16 DWORDs
                | (3u << 25));                             // pad_amount:   4 DWORDs
    g1[1] = (int)(512u << 16);                             // tensor_dim0 = 512 (lo16)
    g1[2] = (int)(512u << 16);                             // tensor_dim1 = 512 (lo16)
    g1[3] = (int)(32u  << 16);                             // tile_dim0 = 32
    g1[4] = 128;                                           // tile_dim1 = 128
    g1[5] = 512;                                           // tensor_dim0_stride = 512
    g1[6] = 0;
    g1[7] = 0;
    i32x4 z4 = {0, 0, 0, 0};
    i32x8 z8 = {0, 0, 0, 0, 0, 0, 0, 0};
    // 6-arg form (clang-23 / therock-10.0 headers)
    __builtin_amdgcn_tensor_load_to_lds(g0, g1, z4, z4, z8, 0);
}

// -------- kernel 4: modulated conv as 9 shifted implicit GEMMs (bf16 WMMA) --------
// WG tile: M=128 (out ch) x N=128 (pixels = 2 image rows). 8 waves in 4(M)x2(N).
// Each wave: 2x4 = eight 16x16 f32 accum tiles; K-step = 32 channels.
#define APITCH 40   // bf16 elements per LDS row (80B): b128 frag reads hit all 64 banks

__global__ __launch_bounds__(256)
void modconv_k(const float* __restrict__ x, const __bf16* __restrict__ wb,
               const float* __restrict__ s, const float* __restrict__ sig,
               float* __restrict__ out)
{
    __shared__ __bf16 As[128 * APITCH];   // A tile: [m][k]   (weights, via TDM)
    __shared__ __bf16 Bs[128 * APITCH];   // B tile: [n][k]   (modulated x)

    const int t      = threadIdx.x;
    const int lane   = t & 31;
    const int wid    = t >> 5;
    const int wave_m = wid >> 1;          // 0..3
    const int wave_n = wid & 1;           // 0..1
    const int b      = blockIdx.z;
    const int o0     = blockIdx.y * 128;
    const int p0     = blockIdx.x * 128;

    const int half = lane >> 4;           // 0/1 : K half selector per ISA layout
    const int lrow = lane & 15;           // M (or N) index within fragment
    const int kb   = half * 8;            // starting K of the low chunk

    const unsigned int As_off = lds_addr_of(&As[0]);

    v8f acc[2][4];
    #pragma unroll
    for (int im = 0; im < 2; ++im)
        #pragma unroll
        for (int in = 0; in < 4; ++in)
            #pragma unroll
            for (int r = 0; r < 8; ++r) acc[im][in][r] = 0.f;

    for (int tap = 0; tap < 9; ++tap) {
        const int dh = tap / 3 - 1;
        const int dw = tap % 3 - 1;
        const __bf16* wtap = wb + (size_t)tap * (CO_ * CI_);

        for (int ic0 = 0; ic0 < CI_; ic0 += 32) {
            __syncthreads();

            // ---- stage A via Tensor Data Mover (one wave issues the DMA) ----
            if (wid == 0)
                tdm_load_A(wtap + (size_t)o0 * CI_ + ic0, As_off);

            // ---- stage B: x * s[b,i] -> bf16, transposed to [n][k], tap shift ----
            {
                int k  = t >> 3;                  // 0..31  channel within chunk
                int ng = t & 7;                   // 16-pixel group
                float sk = s[b * CI_ + ic0 + k];
                const float* xrow = x + (((size_t)b * CI_ + ic0 + k) << 12);
                __builtin_prefetch(xrow + (32 << 12), 0, 0);  // next K-chunk
                #pragma unroll
                for (int j = 0; j < 16; ++j) {
                    int n  = ng * 16 + j;
                    int p  = p0 + n;
                    int hs = (p >> 6) + dh;
                    int wv = (p & 63) + dw;
                    bool ok = ((unsigned)hs < 64u) & ((unsigned)wv < 64u);
                    // wrapped address is always in-bounds: unconditional load,
                    // boundary handled by zeroing the multiplier (no EXEC branches)
                    float v = xrow[((hs & 63) << 6) + (wv & 63)];
                    Bs[n * APITCH + k] = (__bf16)(ok ? v * sk : 0.f);
                }
            }

            if (wid == 0)
                __builtin_amdgcn_s_wait_tensorcnt(0);   // TDM write to LDS done
            __syncthreads();

            // ---- fragments per ISA 7.12.2: A row-per-lane, K split in half-waves ----
            v16bf afrag[2], bfrag[4];
            #pragma unroll
            for (int im = 0; im < 2; ++im) {
                int m = wave_m * 32 + im * 16 + lrow;
                afrag[im] = load_frag16(&As[m * APITCH + kb]);
            }
            #pragma unroll
            for (int in = 0; in < 4; ++in) {
                int n = wave_n * 64 + in * 16 + lrow;
                bfrag[in] = load_frag16(&Bs[n * APITCH + kb]);
            }

            #pragma unroll
            for (int im = 0; im < 2; ++im)
                #pragma unroll
                for (int in = 0; in < 4; ++in)
                    acc[im][in] = __builtin_amdgcn_wmma_f32_16x16x32_bf16(
                        false, afrag[im], false, bfrag[in],
                        (short)0, acc[im][in], false, false);
        }
    }

    // ---- epilogue: scale by sigma_inv and store f32 ----
    // C layout: VGPR r, lanes 0-15 -> M=r, lanes 16-31 -> M=8+r; N = lane&15.
    #pragma unroll
    for (int im = 0; im < 2; ++im) {
        #pragma unroll
        for (int r = 0; r < 8; ++r) {
            int mm = o0 + wave_m * 32 + im * 16 + half * 8 + r;
            float sg = sig[b * CO_ + mm];
            size_t rowbase = ((size_t)(b * CO_ + mm) << 12);
            #pragma unroll
            for (int in = 0; in < 4; ++in) {
                int nn = p0 + wave_n * 64 + in * 16 + lrow;
                out[rowbase + nn] = acc[im][in][r] * sg;
            }
        }
    }
}

// ---------------- launcher ----------------
extern "C" void kernel_launch(void* const* d_in, const int* in_sizes, int n_in,
                              void* d_out, int out_size, void* d_ws, size_t ws_size,
                              hipStream_t stream)
{
    const float* x      = (const float*)d_in[0];   // [16,512,64,64]
    const float* style  = (const float*)d_in[1];   // [16,512]
    const float* weight = (const float*)d_in[2];   // [512,512,3,3]
    const float* aw     = (const float*)d_in[3];   // [512,512]
    const float* ab     = (const float*)d_in[4];   // [512]
    float* out          = (float*)d_out;           // [16,512,64,64]

    char* ws = (char*)d_ws;
    float*  s    = (float*)(ws);                           //  32 KB : [16,512]
    float*  sig  = (float*)(ws + (32 << 10));              //  32 KB : [16,512]
    float*  wsq  = (float*)(ws + (64 << 10));              //   1 MB : [512,512]
    __bf16* wb   = (__bf16*)(ws + (64 << 10) + (1 << 20)); // 4.5 MB : [9,512,512] bf16

    style_affine_k<<<(B_ * CI_) / 256, 256, 0, stream>>>(style, aw, ab, s);
    weight_prep_k <<<(CO_ * CI_) / 256, 256, 0, stream>>>(weight, wb, wsq);
    sigma_k       <<<(B_ * CO_) / 256, 256, 0, stream>>>(wsq, s, sig);

    dim3 grid(P_ / 128, CO_ / 128, B_);   // 32 x 4 x 16
    modconv_k<<<grid, 256, 0, stream>>>(x, wb, s, sig, out);
}
